// RadialField_4080218931366
// MI455X (gfx1250) — matrix operand order repaired and checked by reading, not compiled, not run on hardware
//
#include <hip/hip_runtime.h>
#include <hip/hip_bf16.h>
#include <math.h>

#define NFH 128   // hidden_nf
#define LAYERS 4

typedef __attribute__((ext_vector_type(16))) _Float16 v16h;
typedef __attribute__((ext_vector_type(8)))  float    v8f;

// fast SiLU: x * sigmoid(x), using hw v_exp_f32 + v_rcp_f32 (no IEEE div chain)
__device__ __forceinline__ float fast_silu(float h) {
    return h * __builtin_amdgcn_rcpf(1.0f + __expf(-h));
}
// fast tanh: 1 - 2/(exp(2x)+1); saturates to +/-1 for large |x|
__device__ __forceinline__ float fast_tanh(float y) {
    return 1.0f - 2.0f * __builtin_amdgcn_rcpf(__expf(2.0f * y) + 1.0f);
}

// ---------------------------------------------------------------------------
// init: x = pos, agg = 0, cnt = 0
// ---------------------------------------------------------------------------
__global__ void k_init(const float* __restrict__ pos, float* __restrict__ x,
                       float* __restrict__ agg, float* __restrict__ cnt, int N) {
    int n = blockIdx.x * blockDim.x + threadIdx.x;
    if (n >= N) return;
    x[3 * n + 0] = pos[3 * n + 0];
    x[3 * n + 1] = pos[3 * n + 1];
    x[3 * n + 2] = pos[3 * n + 2];
    agg[3 * n + 0] = 0.0f;
    agg[3 * n + 1] = 0.0f;
    agg[3 * n + 2] = 0.0f;
    cnt[n] = 0.0f;
}

// ---------------------------------------------------------------------------
// degree count over receiving node (row)
// ---------------------------------------------------------------------------
__global__ void k_count(const int* __restrict__ ei, float* __restrict__ cnt, int E) {
    int e = blockIdx.x * blockDim.x + threadIdx.x;
    if (e >= E) return;
    atomicAdd(&cnt[ei[e]], 1.0f);
}

// cnt -> 1 / max(cnt, 1)
__global__ void k_invcnt(float* __restrict__ cnt, int N) {
    int n = blockIdx.x * blockDim.x + threadIdx.x;
    if (n >= N) return;
    float c = cnt[n];
    cnt[n] = __builtin_amdgcn_rcpf(c > 1.0f ? c : 1.0f);
}

// ---------------------------------------------------------------------------
// Edge kernel: one wave32 processes 16 edges.
//   stage A (VALU): h = silu(radial*W1r + ea*W1e + b1), f32, 16 hidden/lane/chunk
//   stage B (WMMA): e_out_raw[16 edges] = h(16x128) . W2(128) via 4x
//                   v_wmma_f32_16x16x32_f16 accumulating along K.
//   A-matrix: W2 chunk in row M=0 (lanes 0 & 16), zeros elsewhere.
//   B-matrix: lane L holds silu(h) for edge (L&15), K = chunk*32 + (L>>4)*16 + t.
//   D VGPR0 lanes 0..15 = dot for the 16 edges -> tanh -> atomic scatter.
// ---------------------------------------------------------------------------
__global__ void k_edge(const float* __restrict__ x,
                       const int*   __restrict__ ei,     // row [0,E), col [E,2E)
                       const float* __restrict__ eattr,  // [E]
                       const float* __restrict__ w1,     // [2, NFH]
                       const float* __restrict__ b1,     // [NFH]
                       const float* __restrict__ w2,     // [NFH]
                       float* __restrict__ agg,
                       int E) {
    __shared__ float sW1r[NFH], sW1e[NFH], sB1[NFH], sW2[NFH];
    int tid = threadIdx.x;
    for (int i = tid; i < NFH; i += blockDim.x) {
        sW1r[i] = w1[i];
        sW1e[i] = w1[NFH + i];
        sB1[i]  = b1[i];
        sW2[i]  = w2[i];
    }
    __syncthreads();

    const int lane  = tid & 31;
    const int wave  = blockIdx.x * (blockDim.x >> 5) + (tid >> 5);
    const int col16 = lane & 15;          // which of the 16 edges this lane serves
    const int khalf = (lane >> 4) << 4;   // 0 or 16: K-half within a 32-chunk
    const int e     = wave * 16 + col16;
    const int eC    = (e < E) ? e : (E - 1);   // clamp; masked at the scatter

    const int r = ei[eC];
    const int c = ei[E + eC];
    const float dx = x[3 * r + 0] - x[3 * c + 0];
    const float dy = x[3 * r + 1] - x[3 * c + 1];
    const float dz = x[3 * r + 2] - x[3 * c + 2];
    const float radial = sqrtf(dx * dx + dy * dy + dz * dz);
    const float ea = eattr[eC];

    v8f acc = {0.f, 0.f, 0.f, 0.f, 0.f, 0.f, 0.f, 0.f};

#pragma unroll
    for (int ch = 0; ch < 4; ++ch) {
        const int kb = ch * 32 + khalf;
        // stage A: 16 hidden activations for (edge, K-half) on the VALU/TRANS pipe
        v16h b;
#pragma unroll
        for (int t = 0; t < 16; ++t) {
            const int k = kb + t;
            float h = fmaf(radial, sW1r[k], fmaf(ea, sW1e[k], sB1[k]));
            b[t] = (_Float16)fast_silu(h);
        }
        // A-matrix: W2 chunk lives in row M=0 only (lanes 0 and 16), zeros elsewhere
        v16h a = {};
        if (lane == 0 || lane == 16) {
            const int ab = ch * 32 + (lane ? 16 : 0);
#pragma unroll
            for (int t = 0; t < 8; ++t) {
                a[t]     = (_Float16)sW2[ab + t];
                a[t + 8] = (_Float16)sW2[ab + 8 + t];
            }
        }
        acc = __builtin_amdgcn_wmma_f32_16x16x32_f16(
                  false, a, false, b, (short)0, acc, false, false);
    }

    // D row M=0 -> VGPR0, lanes 0..15 hold the 16 edge dots
    if (lane < 16 && e < E) {
        const float t = fast_tanh(acc[0]);
        atomicAdd(&agg[3 * r + 0], dx * t);
        atomicAdd(&agg[3 * r + 1], dy * t);
        atomicAdd(&agg[3 * r + 2], dz * t);
    }
}

// ---------------------------------------------------------------------------
// Node kernel: x += agg/max(cnt,1) + v * coord_mlp_vel(||v||); re-zero agg for
// the next layer (first layer's zero comes from k_init).
// ---------------------------------------------------------------------------
__global__ void k_node(float* __restrict__ x, const float* __restrict__ v,
                       float* __restrict__ agg, const float* __restrict__ invcnt,
                       const float* __restrict__ vw1, const float* __restrict__ vb1,
                       const float* __restrict__ vw2, const float* __restrict__ vb2,
                       int N) {
    int n = blockIdx.x * blockDim.x + threadIdx.x;
    if (n >= N) return;
    const float vx = v[3 * n + 0], vy = v[3 * n + 1], vz = v[3 * n + 2];
    const float vn = sqrtf(vx * vx + vy * vy + vz * vz);
    float acc = vb2[0];
#pragma unroll 4
    for (int j = 0; j < NFH; ++j) {
        float h = fmaf(vn, vw1[j], vb1[j]);
        acc = fmaf(fast_silu(h), vw2[j], acc);
    }
    const float ic = invcnt[n];
    x[3 * n + 0] += agg[3 * n + 0] * ic + vx * acc;
    x[3 * n + 1] += agg[3 * n + 1] * ic + vy * acc;
    x[3 * n + 2] += agg[3 * n + 2] * ic + vz * acc;
    agg[3 * n + 0] = 0.0f;
    agg[3 * n + 1] = 0.0f;
    agg[3 * n + 2] = 0.0f;
}

// ---------------------------------------------------------------------------
extern "C" void kernel_launch(void* const* d_in, const int* in_sizes, int n_in,
                              void* d_out, int out_size, void* d_ws, size_t ws_size,
                              hipStream_t stream) {
    const float* pos    = (const float*)d_in[0];
    const float* v      = (const float*)d_in[1];
    const float* eattr  = (const float*)d_in[2];
    const float* phi_w1 = (const float*)d_in[3];  // [L, 2, NFH]
    const float* phi_b1 = (const float*)d_in[4];  // [L, NFH]
    const float* phi_w2 = (const float*)d_in[5];  // [L, NFH, 1]
    const float* vel_w1 = (const float*)d_in[6];  // [L, 1, NFH]
    const float* vel_b1 = (const float*)d_in[7];  // [L, NFH]
    const float* vel_w2 = (const float*)d_in[8];  // [L, NFH, 1]
    const float* vel_b2 = (const float*)d_in[9];  // [L, 1]
    const int*   ei     = (const int*)d_in[10];   // [2, E]

    const int N = in_sizes[0] / 3;
    const int E = in_sizes[2];        // edge_attr has EA=1 -> E elements

    float* x   = (float*)d_out;       // working coordinates, also the output
    float* agg = (float*)d_ws;        // [N,3]
    float* cnt = agg + (size_t)3 * N; // [N] degree -> 1/max(deg,1)

    const int TB = 256;
    const int nBlkN = (N + TB - 1) / TB;
    const int nBlkE = (E + TB - 1) / TB;
    const int waves = (E + 15) / 16;                 // one wave32 per 16 edges
    const int nBlkEdge = (waves + (TB / 32) - 1) / (TB / 32);

    k_init  <<<nBlkN, TB, 0, stream>>>(pos, x, agg, cnt, N);
    k_count <<<nBlkE, TB, 0, stream>>>(ei, cnt, E);
    k_invcnt<<<nBlkN, TB, 0, stream>>>(cnt, N);

    for (int l = 0; l < LAYERS; ++l) {
        k_edge<<<nBlkEdge, TB, 0, stream>>>(
            x, ei, eattr,
            phi_w1 + (size_t)l * 2 * NFH,
            phi_b1 + (size_t)l * NFH,
            phi_w2 + (size_t)l * NFH,
            agg, E);
        k_node<<<nBlkN, TB, 0, stream>>>(
            x, v, agg, cnt,
            vel_w1 + (size_t)l * NFH,
            vel_b1 + (size_t)l * NFH,
            vel_w2 + (size_t)l * NFH,
            vel_b2 + (size_t)l,
            N);
    }
}
